// StageNet_59047210386105
// MI455X (gfx1250) — matrix operand order repaired and checked by reading, not compiled, not run on hardware
//
#include <hip/hip_runtime.h>
#include <hip/hip_bf16.h>
#include <math.h>

typedef __attribute__((ext_vector_type(16))) _Float16 v16h;
typedef __attribute__((ext_vector_type(8)))  float    v8f;

#define C_  32
#define D_  48
#define H_  128
#define W_  128
#define HW_ (H_ * W_)
#define G_  8
#define NVIEWS 3.0f
#define INV_SQRT_C 0.17677669529663687f   // 1/sqrt(32)
#define INV_ATTN   0.5f                    // 1/ATTN_TEMP

// ---------------------------------------------------------------------------
// Setup: fuse projections, compute rot/trans of src_new @ inv(ref_new).
// proj layout: (B=1, V=3, 2, 4, 4); elem (v,j,r,c) at v*32 + j*16 + r*4 + c.
// ws: per view i in {0,1}: 9 floats rot (row-major) + 3 floats trans.
// ---------------------------------------------------------------------------
__global__ void setup_proj(const float* __restrict__ proj, float* __restrict__ ws) {
    if (threadIdx.x != 0 || blockIdx.x != 0) return;
    float F[3][12];
    for (int v = 0; v < 3; ++v) {
        const float* E = proj + v * 32;        // extrinsic (4x4), row3=[0,0,0,1]
        const float* K = proj + v * 32 + 16;   // intrinsic (4x4)
        for (int r = 0; r < 3; ++r)
            for (int c = 0; c < 4; ++c) {
                float s = 0.f;
                for (int k = 0; k < 3; ++k) s += K[r*4+k] * E[k*4+c];
                F[v][r*4+c] = s;
            }
    }
    // ref fused: [[A, b],[0,1]]  ->  inv = [[Ai, -Ai b],[0,1]]
    float a[9];
    for (int r = 0; r < 3; ++r)
        for (int c = 0; c < 3; ++c) a[r*3+c] = F[0][r*4+c];
    float b0 = F[0][3], b1 = F[0][7], b2 = F[0][11];
    float c00 =  (a[4]*a[8] - a[5]*a[7]);
    float c01 = -(a[3]*a[8] - a[5]*a[6]);
    float c02 =  (a[3]*a[7] - a[4]*a[6]);
    float det = a[0]*c00 + a[1]*c01 + a[2]*c02;
    float id  = 1.f / det;
    float Ai[9];
    Ai[0] =  c00 * id;
    Ai[1] = -(a[1]*a[8] - a[2]*a[7]) * id;
    Ai[2] =  (a[1]*a[5] - a[2]*a[4]) * id;
    Ai[3] =  c01 * id;
    Ai[4] =  (a[0]*a[8] - a[2]*a[6]) * id;
    Ai[5] = -(a[0]*a[5] - a[2]*a[3]) * id;
    Ai[6] =  c02 * id;
    Ai[7] = -(a[0]*a[7] - a[1]*a[6]) * id;
    Ai[8] =  (a[0]*a[4] - a[1]*a[3]) * id;
    for (int i = 0; i < 2; ++i) {
        const float* Fs = F[i+1];
        float R[9];
        for (int r = 0; r < 3; ++r)
            for (int c = 0; c < 3; ++c) {
                float s = 0.f;
                for (int k = 0; k < 3; ++k) s += Fs[r*4+k] * Ai[k*3+c];
                R[r*3+c] = s;
            }
        float t0 = Fs[3]  - (R[0]*b0 + R[1]*b1 + R[2]*b2);
        float t1 = Fs[7]  - (R[3]*b0 + R[4]*b1 + R[5]*b2);
        float t2 = Fs[11] - (R[6]*b0 + R[7]*b1 + R[8]*b2);
        float* o = ws + i * 12;
        for (int k = 0; k < 9; ++k) o[k] = R[k];
        o[9] = t0; o[10] = t1; o[11] = t2;
    }
}

// Per-pixel LDS state (one wave32 == one pixel)
struct PixLDS {
    float cor[2][D_][G_];  // per-view grouped correlation; cor[0] becomes accumulator
    float wsum[D_];
    float varc[D_];
    float sm[D_];
};

__device__ __forceinline__ void cook_corner(float xc, float yc, float wgt,
                                            int& off, float& wo) {
    bool v = (xc >= 0.f) && (xc <= (float)(W_ - 1)) &&
             (yc >= 0.f) && (yc <= (float)(H_ - 1));
    float xq = fminf(fmaxf(xc, 0.f), (float)(W_ - 1));
    float yq = fminf(fmaxf(yc, 0.f), (float)(H_ - 1));
    off = ((int)yq) * W_ + (int)xq;
    wo  = v ? wgt : 0.f;
}

// ---------------------------------------------------------------------------
// Main fused kernel: 1 wave32 per pixel, 4 waves (128 threads) per block.
// ---------------------------------------------------------------------------
__global__ __launch_bounds__(128)
void stage_main(const float* __restrict__ ref,   const float* __restrict__ src,
                const float* __restrict__ dh,    const float* __restrict__ wsim,
                const float* __restrict__ wvar,  const float* __restrict__ tmask,
                const float* __restrict__ rt,    float* __restrict__ out) {
    __shared__ PixLDS lds[4];
    const int lane = threadIdx.x & 31;
    const int wid  = threadIdx.x >> 5;
    const int pix  = blockIdx.x * 4 + wid;
    PixLDS& P = lds[wid];

    const int   h  = pix >> 7, w = pix & (W_ - 1);
    const float xf = (float)w, yf = (float)h;

    // Per-view rot*[x,y,1] bases and translations (uniform per pixel)
    float bx[2], by[2], bz[2], tx[2], ty[2], tz[2];
    #pragma unroll
    for (int v = 0; v < 2; ++v) {
        const float* m = rt + v * 12;
        bx[v] = m[0]*xf + m[1]*yf + m[2];
        by[v] = m[3]*xf + m[4]*yf + m[5];
        bz[v] = m[6]*xf + m[7]*yf + m[8];
        tx[v] = m[9]; ty[v] = m[10]; tz[v] = m[11];
    }

    const int  m16  = lane & 15;
    const bool lo   = lane < 16;
    const int  kadd = lo ? 0 : 8;

    // A-matrix lane->channel map per ISA 16-bit A 16x32 layout
    int   chans[16];
    float refv[16], wv[16];
    #pragma unroll
    for (int j = 0; j < 16; ++j) {
        int c = ((j < 8) ? j : j + 8) + kadd;
        chans[j] = c;
        refv[j]  = ref[c * HW_ + pix];
        wv[j]    = wvar[c];
    }

    // B: 32x16 block-diagonal (ref/4 over each 4-channel group), per B layout:
    // lane n holds K=0..7,16..23 (lanes 0-15) or K=8..15,24..31 (lanes 16-31) of column n.
    v16h Bm;
    {
        int n = m16, kbase = lo ? 0 : 8;
        #pragma unroll
        for (int j = 0; j < 16; ++j) {
            int c = ((j < 8) ? j : j + 8) + kbase;
            float bvv = ((c >> 2) == n) ? ref[c * HW_ + pix] * 0.25f : 0.f;
            Bm[j] = (_Float16)bvv;
        }
    }
    v8f cz = {};

    // ------------------ Phase 1: warp + WMMA correlation + variance --------
    for (int t = 0; t < 3; ++t) {
        const int   d   = t * 16 + m16;
        const float dep = dh[d * HW_ + pix];

        float svals[2][16];
        v16h  Amat[2];
        #pragma unroll
        for (int v = 0; v < 2; ++v) {
            float pz  = bz[v] * dep + tz[v];
            float izv = 1.f / pz;
            float px  = (bx[v] * dep + tx[v]) * izv;
            float py  = (by[v] * dep + ty[v]) * izv;
            float x0f = floorf(px), y0f = floorf(py);
            float wx1 = px - x0f, wx0 = 1.f - wx1;
            float wy1 = py - y0f, wy0 = 1.f - wy1;
            int   o00, o10, o01, o11;
            float w00, w10, w01, w11;
            cook_corner(x0f,       y0f,       wx0 * wy0, o00, w00);
            cook_corner(x0f + 1.f, y0f,       wx1 * wy0, o10, w10);
            cook_corner(x0f,       y0f + 1.f, wx0 * wy1, o01, w01);
            cook_corner(x0f + 1.f, y0f + 1.f, wx1 * wy1, o11, w11);
            const float* sb = src + v * C_ * HW_;
            #pragma unroll
            for (int j = 0; j < 16; ++j) {
                const float* fp = sb + chans[j] * HW_;
                float s = w00 * fp[o00] + w10 * fp[o10] +
                          w01 * fp[o01] + w11 * fp[o11];
                svals[v][j] = s;
                Amat[v][j]  = (_Float16)s;
            }
        }

        // Grouped correlation: D[16 depths x 8 groups] via 16x16x32 f16 WMMA
        #pragma unroll
        for (int v = 0; v < 2; ++v) {
            v8f cc = __builtin_amdgcn_wmma_f32_16x16x32_f16(
                false, Amat[v], false, Bm, (short)0, cz, false, false);
            if (m16 < G_) {
                int g = m16;
                int dbase = t * 16 + (lo ? 0 : 8);
                #pragma unroll
                for (int r = 0; r < 8; ++r) P.cor[v][dbase + r][g] = cc[r];
            }
        }

        // Variance dot: each lane covers 16 channels of its depth, pair via xor-16
        float p = 0.f;
        #pragma unroll
        for (int j = 0; j < 16; ++j) {
            float s1 = svals[0][j], s2 = svals[1][j], rr = refv[j];
            float vs = (rr + s1 + s2) * (1.f / NVIEWS);
            float vq = (rr * rr + s1 * s1 + s2 * s2) * (1.f / NVIEWS);
            p += wv[j] * (vq - vs * vs);
        }
        p += __shfl_xor(p, 16, 32);
        if (lo) P.varc[t * 16 + m16] = p;
    }
    __syncthreads();

    // ------------------ Phase 2: per-view attention softmax over depth -----
    int dm[2] = { lane, lane + 32 };
    const int nd = lo ? 2 : 1;
    for (int v = 0; v < 2; ++v) {
        for (int q = 0; q < nd; ++q) {
            int dd = dm[q];
            float s = 0.f;
            #pragma unroll
            for (int g = 0; g < G_; ++g) s += P.cor[v][dd][g];
            P.sm[dd] = s * INV_ATTN;
        }
        __syncthreads();
        float mx = -1e30f;
        for (int dd = 0; dd < D_; ++dd) mx = fmaxf(mx, P.sm[dd]);
        float den = 0.f;
        for (int dd = 0; dd < D_; ++dd) den += __expf(P.sm[dd] - mx);
        float sc = INV_SQRT_C / den;
        for (int q = 0; q < nd; ++q) {
            int dd = dm[q];
            float wgt = __expf(P.sm[dd] - mx) * sc;
            if (v == 0) {
                P.wsum[dd] = 1e-8f + wgt;
                #pragma unroll
                for (int g = 0; g < G_; ++g) P.cor[0][dd][g] *= wgt;
            } else {
                P.wsum[dd] += wgt;
                #pragma unroll
                for (int g = 0; g < G_; ++g)
                    P.cor[0][dd][g] += wgt * P.cor[1][dd][g];
            }
        }
        __syncthreads();
    }

    // ------------------ Phase 3: cost regs, softmaxes, blend, argmax -------
    for (int q = 0; q < nd; ++q) {
        int dd = dm[q];
        float cr = 0.f;
        #pragma unroll
        for (int g = 0; g < G_; ++g) cr += P.cor[0][dd][g] * wsim[g];
        P.sm[dd] = cr / P.wsum[dd];
    }
    __syncthreads();
    float m1 = -1e30f, m2 = -1e30f;
    for (int dd = 0; dd < D_; ++dd) {
        m1 = fmaxf(m1, P.sm[dd]);
        m2 = fmaxf(m2, P.varc[dd]);
    }
    float den1 = 0.f, den2 = 0.f;
    for (int dd = 0; dd < D_; ++dd) {
        den1 += __expf(P.sm[dd]   - m1);
        den2 += __expf(P.varc[dd] - m2);
    }
    const float msk = tmask[pix];
    float* probOut = out + 4 * HW_;
    for (int q = 0; q < nd; ++q) {
        int dd = dm[q];
        float sp = __expf(P.sm[dd]   - m1) / den1;
        float vp = __expf(P.varc[dd] - m2) / den2;
        float pb = (1.f - msk) * sp + msk * vp;
        probOut[dd * HW_ + pix] = pb;
        P.cor[1][dd][0] = sp;   // reuse freed cor[1] as scratch
        P.cor[1][dd][1] = vp;
        P.cor[1][dd][2] = pb;
    }
    __syncthreads();
    if (lane == 0) {
        float bp = -1.f, msp = -1.f, mvp = -1.f;
        int bi = 0;
        for (int dd = 0; dd < D_; ++dd) {
            float pb = P.cor[1][dd][2];
            if (pb > bp) { bp = pb; bi = dd; }
            msp = fmaxf(msp, P.cor[1][dd][0]);
            mvp = fmaxf(mvp, P.cor[1][dd][1]);
        }
        float depth = dh[bi * HW_ + pix];
        float itv   = 1.f / dh[2 * HW_ + pix] - 1.f / dh[1 * HW_ + pix];
        out[pix]            = depth;      // depth
        out[HW_ + pix]      = bp;         // photometric_confidence
        out[2 * HW_ + pix]  = mvp;        // var_photometric_confidence
        out[3 * HW_ + pix]  = msp;        // sim_photometric_confidence
        out[52 * HW_ + pix] = 1.f / depth + itv;  // inverse_min_depth
        out[53 * HW_ + pix] = 1.f / depth - itv;  // inverse_max_depth
    }
}

extern "C" void kernel_launch(void* const* d_in, const int* in_sizes, int n_in,
                              void* d_out, int out_size, void* d_ws, size_t ws_size,
                              hipStream_t stream) {
    const float* ref   = (const float*)d_in[0];
    const float* src   = (const float*)d_in[1];
    const float* proj  = (const float*)d_in[2];
    const float* dh    = (const float*)d_in[3];
    const float* wsim  = (const float*)d_in[4];
    const float* wvar  = (const float*)d_in[5];
    const float* tmask = (const float*)d_in[6];
    float* out = (float*)d_out;
    float* rt  = (float*)d_ws;  // 24 floats: rot(9)+trans(3) per source view

    setup_proj<<<dim3(1), dim3(1), 0, stream>>>(proj, rt);
    stage_main<<<dim3(HW_ / 4), dim3(128), 0, stream>>>(
        ref, src, dh, wsim, wvar, tmask, rt, out);
}
